// cNet_DVS_86303072846335
// MI455X (gfx1250) — matrix operand order repaired and checked by reading, not compiled, not run on hardware
//
#include <hip/hip_runtime.h>

// ---------------------------------------------------------------------------
// Spiking CNN scan (T=300, B=32) fully fused into ONE persistent kernel.
//
// The scan is serial in T, so the whole problem is latency-bound
// (~106 MFLOP/step). One 1024-thread workgroup (32 wave32 waves on one WGP)
// runs all 300 steps; layers are separated by workgroup barriers. Weights are
// staged in LDS as f16 B-transposed matrices; LIF membranes (f32) and spike
// maps (f16) live in L2-resident global workspace. Convs are implicit GEMMs
// on v_wmma_f32_16x16x32_f16 with f32 accumulation (spikes are exactly 0/1,
// so f16 inputs are numerically safe).
// ---------------------------------------------------------------------------

typedef __attribute__((ext_vector_type(16))) _Float16 v16h;
typedef __attribute__((ext_vector_type(8)))  _Float16 v8h;
typedef __attribute__((ext_vector_type(8)))  float    v8f;

// ---- problem constants ----
constexpr int Bn = 32;    // batch
constexpr int Tn = 300;   // timesteps

// per-layer state sizes ([Cout][B*OH*OW] layout)
constexpr int NS1 = 16 * 7200;   // 16 x (32*15*15)
constexpr int NS2 = 32 * 5408;   // 32 x (32*13*13)
constexpr int NS3 = 64 * 1152;   // 64 x (32*6*6)
constexpr int NS4 = 10 * 288;    // 10 x (32*3*3)

// ---- workspace layout (all byte offsets stay 16B aligned) ----
constexpr size_t NF32   = 2ull * (NS1 + NS2 + NS3 + NS4);         // m1+m2, f32
constexpr size_t OFF_S  = NF32 * 4;                               // spikes, f16
constexpr size_t NSP    = (size_t)NS1 + NS2 + NS3 + NS4;
constexpr size_t OFF_BT4 = OFF_S  + NSP * 2;                      // f16 [16][1024]
constexpr size_t OFF_A1  = OFF_BT4 + (size_t)16 * 1024 * 2;       // f16 [7200][32]
constexpr size_t OFF_A2  = OFF_A1  + (size_t)7200 * 32 * 2;       // f16 [5408][160]
constexpr size_t OFF_A3  = OFF_A2  + (size_t)5408 * 160 * 2;      // f16 [1152][288]
constexpr size_t OFF_A4  = OFF_A3  + (size_t)1152 * 288 * 2;      // f16 [288][1024]
constexpr size_t OFF_A5  = OFF_A4  + (size_t)288 * 1024 * 2;      // f16 [32][96]

// ---------------------------------------------------------------------------
// Implicit-GEMM + fused LIF update for one conv layer.
//   A  : [MTOT][KPAD] f16 im2col patches (global, 16B-aligned rows)
//   Bt : [ceil16(CO)][KPAD] f16 transposed weights (LDS or global)
//   state m1/m2 (f32), sp (f16) laid out [CO][MTOT]
// Fragment layouts per CDNA5 ISA 7.12.2 (16-bit A 16x32, B 32x16, f32 C).
// ---------------------------------------------------------------------------
template<int MTOT, int KPAD, int KCH, int CO>
__device__ __forceinline__ void gemm_lif(const _Float16* __restrict__ A,
                                         const _Float16* __restrict__ Bt,
                                         float* __restrict__ m1,
                                         float* __restrict__ m2,
                                         _Float16* __restrict__ sp,
                                         int wave, int lane)
{
  constexpr int NT = (CO + 15) / 16;
  constexpr int MT = MTOT / 16;
  const int l15 = lane & 15;
  const int lh  = lane >> 4;     // 0: low half of wave, 1: high half

  for (int tile = wave; tile < MT * NT; tile += 32) {
    const int mt = tile / NT;
    const int nt = tile - mt * NT;
    const _Float16* Arow = A  + (mt * 16 + l15) * KPAD + (lh ? 8  : 0);
    const _Float16* Brow = Bt + (nt * 16 + l15) * KPAD + (lh ? 8  : 0);

    // prefetch next tile's A row (global_prefetch_b8) — wave-uniform branch
    if (tile + 32 < MT * NT) {
      const int tn = tile + 32;
      __builtin_prefetch(A + ((tn / NT) * 16 + l15) * KPAD, 0, 1);
    }

    v8f c = {};
#pragma unroll 4
    for (int kc = 0; kc < KCH; ++kc) {
      const int k0 = kc * 32;
      // A lane<16: K{k0..k0+7, k0+16..k0+23}; lane>=16: shifted by 8
      v8h alo = *(const v8h*)(Arow + k0);
      v8h ahi = *(const v8h*)(Arow + k0 + 16);
      // B lane<16: Bt[n][k0..k0+7, k0+16..k0+23]; lane>=16: shifted by 8
      v8h blo = *(const v8h*)(Brow + k0);
      v8h bhi = *(const v8h*)(Brow + k0 + 16);
      v16h a = __builtin_shufflevector(alo, ahi, 0,1,2,3,4,5,6,7,8,9,10,11,12,13,14,15);
      v16h b = __builtin_shufflevector(blo, bhi, 0,1,2,3,4,5,6,7,8,9,10,11,12,13,14,15);
      c = __builtin_amdgcn_wmma_f32_16x16x32_f16(false, a, false, b,
                                                 (short)0, c, false, false);
    }

    // C layout: VGPR r, lane l -> (M = r + 8*(l>=16), N = l&15)
    const int col = nt * 16 + l15;
    const int mb  = mt * 16 + lh * 8;
    if (col < CO) {
#pragma unroll
      for (int r = 0; r < 8; ++r) {
        const int m   = mb + r;
        const int idx = col * MTOT + m;
        float aa = m1[idx], bb = m2[idx];
        float s0 = (float)sp[idx];
        aa = aa * 0.8f * (1.0f - s0) + c[r];          // m1*DECAY1*(1-sp)+cur
        bb = bb * 0.6f + 0.1f * s0;                   // m2*DECAY2+CONST*sp
        const float ns = (aa - bb - 0.5f) > 0.0f ? 1.0f : 0.0f;
        m1[idx] = aa; m2[idx] = bb; sp[idx] = (_Float16)ns;
      }
    }
  }
}

// ---------------------------------------------------------------------------
// im2col: spike map [CI][B*IH*IW] (f16) -> patch matrix [MTOT][KPAD] (f16)
// m = nb*OH*OW + oh*OW + ow ; k = ci*KH*KW + kh*KW + kw ; pad K..KPAD with 0.
// ---------------------------------------------------------------------------
template<int MTOT,int CI,int KH,int KW,int OH,int OW,int IH,int IW,int STR,int KPAD>
__device__ __forceinline__ void im2col(_Float16* __restrict__ A,
                                       const _Float16* __restrict__ sin_,
                                       int tid)
{
  constexpr int K    = CI * KH * KW;
  constexpr int ROWS = MTOT * CI * KH;
  for (int e = tid; e < ROWS; e += 1024) {
    const int kh  = e % KH;
    const int t1  = e / KH;
    const int ci  = t1 % CI;
    const int m   = t1 / CI;
    const int nb  = m / (OH * OW);
    const int rem = m - nb * (OH * OW);
    const int oh  = rem / OW;
    const int ow  = rem - oh * OW;
    const _Float16* src = sin_ + ci * (Bn * IH * IW) + nb * (IH * IW)
                        + (oh * STR + kh) * IW + ow * STR;
    _Float16* dst = A + m * KPAD + ci * (KH * KW) + kh * KW;
#pragma unroll
    for (int kw = 0; kw < KW; ++kw) dst[kw] = src[kw];
  }
  if constexpr (KPAD > K) {
    constexpr int PAD = KPAD - K;
    for (int e = tid; e < MTOT * PAD; e += 1024) {
      const int m = e / PAD;
      A[m * KPAD + K + (e - m * PAD)] = (_Float16)0.0f;
    }
  }
}

// ---------------------------------------------------------------------------
__global__ __launch_bounds__(1024, 1)
void snn_scan_kernel(const float* __restrict__ x,   // [32,300,2,32,32]
                     const float* __restrict__ w1,  // [16,2,4,4]
                     const float* __restrict__ w2,  // [32,16,3,3]
                     const float* __restrict__ w3,  // [64,32,3,3]
                     const float* __restrict__ w4,  // [10,64,4,4]
                     const float* __restrict__ wfc, // [11,90]
                     float* __restrict__ out,       // [32,11]
                     char* __restrict__ ws)
{
  // LDS: transposed f16 weights + FC state + spike accumulator (~58 KB)
  __shared__ __align__(16) _Float16 sBt1[16 * 32];
  __shared__ __align__(16) _Float16 sBt2[32 * 160];
  __shared__ __align__(16) _Float16 sBt3[64 * 288];
  __shared__ __align__(16) _Float16 sBtf[16 * 96];
  __shared__ float sM1f[512], sM2f[512], sSf[512], sAcc[512];

  const int tid  = threadIdx.x;
  const int wave = tid >> 5;
  const int lane = tid & 31;

  float*   m1_1 = (float*)ws;
  float*   m2_1 = m1_1 + NS1;
  float*   m1_2 = m2_1 + NS1;
  float*   m2_2 = m1_2 + NS2;
  float*   m1_3 = m2_2 + NS2;
  float*   m2_3 = m1_3 + NS3;
  float*   m1_4 = m2_3 + NS3;
  float*   m2_4 = m1_4 + NS4;
  _Float16* s1  = (_Float16*)(ws + OFF_S);
  _Float16* s2  = s1 + NS1;
  _Float16* s3  = s2 + NS2;
  _Float16* s4  = s3 + NS3;
  _Float16* Bt4 = (_Float16*)(ws + OFF_BT4);
  _Float16* A1  = (_Float16*)(ws + OFF_A1);
  _Float16* A2  = (_Float16*)(ws + OFF_A2);
  _Float16* A3  = (_Float16*)(ws + OFF_A3);
  _Float16* A4  = (_Float16*)(ws + OFF_A4);
  _Float16* A5  = (_Float16*)(ws + OFF_A5);

  // ---- one-time init: stage weights (f16, Bt layout), zero state ----
  for (int e = tid; e < 16 * 32; e += 1024) sBt1[e] = (_Float16)w1[e];
  for (int e = tid; e < 32 * 160; e += 1024) {
    const int co = e / 160, k = e - co * 160;
    sBt2[e] = (_Float16)(k < 144 ? w2[co * 144 + k] : 0.0f);
  }
  for (int e = tid; e < 64 * 288; e += 1024) sBt3[e] = (_Float16)w3[e];
  for (int e = tid; e < 16 * 96; e += 1024) {
    const int co = e / 96, k = e - co * 96;
    sBtf[e] = (_Float16)((co < 11 && k < 90) ? wfc[co * 90 + k] : 0.0f);
  }
  for (int e = tid; e < 512; e += 1024) { sM1f[e]=0.f; sM2f[e]=0.f; sSf[e]=0.f; sAcc[e]=0.f; }
  for (int e = tid; e < 16 * 1024; e += 1024) {
    const int co = e >> 10;
    Bt4[e] = (_Float16)(co < 10 ? w4[e] : 0.0f);
  }
  { // zero membranes (f32) and spikes (f16)
    float* f32b = (float*)ws;
    for (size_t e = tid; e < NF32; e += 1024) f32b[e] = 0.0f;
    for (size_t e = tid; e < NSP; e += 1024) s1[e] = (_Float16)0.0f;
  }
  __syncthreads();

  // ---- the serial time scan ----
  for (int t = 0; t < Tn; ++t) {
    // A1: clip(x)+transpose(H<->W) patches. k = ci*16+kh*4+kw, KPAD=32.
    for (int e = tid; e < 7200 * 2 * 4; e += 1024) {
      const int kh = e & 3;
      const int t1 = e >> 2;
      const int ci = t1 & 1;
      const int m  = t1 >> 1;
      const int nb = m / 225, rem = m - nb * 225;
      const int oh = rem / 15, ow = rem - oh * 15;
      // xt[b,c,i,j] = x[b,t,c,j,i]; conv reads xt[b,ci,oh*2+kh,ow*2+kw]
      const float* src = x + (((size_t)nb * Tn + t) * 2 + ci) * 1024
                           + (ow * 2) * 32 + (oh * 2 + kh);
      _Float16* dst = A1 + m * 32 + ci * 16 + kh * 4;
#pragma unroll
      for (int kw = 0; kw < 4; ++kw) {
        float v = src[kw * 32];
        v = fminf(fmaxf(v, 0.0f), 1.0f);
        dst[kw] = (_Float16)v;
      }
    }
    __syncthreads();
    gemm_lif<7200, 32, 1, 16>(A1, sBt1, m1_1, m2_1, s1, wave, lane);   // conv1
    __syncthreads();
    im2col<5408, 16, 3, 3, 13, 13, 15, 15, 1, 160>(A2, s1, tid);
    __syncthreads();
    gemm_lif<5408, 160, 5, 32>(A2, sBt2, m1_2, m2_2, s2, wave, lane);  // conv2
    __syncthreads();
    im2col<1152, 32, 3, 3, 6, 6, 13, 13, 2, 288>(A3, s2, tid);
    __syncthreads();
    gemm_lif<1152, 288, 9, 64>(A3, sBt3, m1_3, m2_3, s3, wave, lane);  // conv3
    __syncthreads();
    im2col<288, 64, 4, 4, 3, 3, 6, 6, 1, 1024>(A4, s3, tid);
    __syncthreads();
    gemm_lif<288, 1024, 32, 10>(A4, Bt4, m1_4, m2_4, s4, wave, lane);  // conv4
    __syncthreads();

    // A5: s4.reshape(B,90), k = co*9 + pix, KPAD=96
    for (int e = tid; e < 32 * 96; e += 1024) {
      const int b = e / 96, k = e - b * 96;
      _Float16 v = (_Float16)0.0f;
      if (k < 90) {
        const int co = k / 9, pix = k - co * 9;
        v = s4[co * 288 + b * 9 + pix];
      }
      A5[e] = v;
    }
    __syncthreads();

    // FC (M=32, K=96, N padded 11->16): waves 0,1 each do one 16x16 tile
    if (wave < 2) {
      const int l15 = lane & 15, lh = lane >> 4;
      const _Float16* Arow = A5   + (wave * 16 + l15) * 96 + (lh ? 8  : 0);
      const _Float16* Brow = sBtf + l15 * 96 + (lh ? 8  : 0);
      v8f c = {};
#pragma unroll
      for (int kc = 0; kc < 3; ++kc) {
        const int k0 = kc * 32;
        v8h alo = *(const v8h*)(Arow + k0);
        v8h ahi = *(const v8h*)(Arow + k0 + 16);
        v8h blo = *(const v8h*)(Brow + k0);
        v8h bhi = *(const v8h*)(Brow + k0 + 16);
        v16h a = __builtin_shufflevector(alo, ahi, 0,1,2,3,4,5,6,7,8,9,10,11,12,13,14,15);
        v16h b = __builtin_shufflevector(blo, bhi, 0,1,2,3,4,5,6,7,8,9,10,11,12,13,14,15);
        c = __builtin_amdgcn_wmma_f32_16x16x32_f16(false, a, false, b,
                                                   (short)0, c, false, false);
      }
      const int col = l15;
      const int mb  = wave * 16 + lh * 8;
      if (col < 11) {
#pragma unroll
        for (int r = 0; r < 8; ++r) {
          const int b   = mb + r;
          const int idx = col * 32 + b;
          float aa = sM1f[idx], bb = sM2f[idx], s0 = sSf[idx];
          aa = aa * 0.8f * (1.0f - s0) + c[r];
          bb = bb * 0.6f + 0.1f * s0;
          const float ns = (aa - bb - 0.5f) > 0.0f ? 1.0f : 0.0f;
          sM1f[idx] = aa; sM2f[idx] = bb; sSf[idx] = ns;
          sAcc[idx] += ns;
        }
      }
    }
    __syncthreads();
  }

  // out[b,co] = acc/T
  for (int e = tid; e < 32 * 11; e += 1024) {
    const int b = e / 11, co = e - b * 11;
    out[e] = sAcc[co * 32 + b] * (1.0f / 300.0f);
  }
}

// ---------------------------------------------------------------------------
extern "C" void kernel_launch(void* const* d_in, const int* in_sizes, int n_in,
                              void* d_out, int out_size, void* d_ws, size_t ws_size,
                              hipStream_t stream)
{
  (void)in_sizes; (void)n_in; (void)out_size; (void)ws_size;
  const float* x   = (const float*)d_in[0];
  const float* w1  = (const float*)d_in[1];
  const float* w2  = (const float*)d_in[2];
  const float* w3  = (const float*)d_in[3];
  const float* w4  = (const float*)d_in[4];
  const float* wfc = (const float*)d_in[5];
  snn_scan_kernel<<<1, 1024, 0, stream>>>(x, w1, w2, w3, w4, wfc,
                                          (float*)d_out, (char*)d_ws);
}
